// NeuralODEWrapper_6906307412138
// MI455X (gfx1250) — matrix operand order repaired and checked by reading, not compiled, not run on hardware
//
#include <hip/hip_runtime.h>

// Neural ODE (RK4, 100 steps) for MI455X / gfx1250.
// Each workgroup (256 threads = 8 waves, wave32) integrates 16 batch rows for
// all 100 steps. Weight matrices are held as WMMA B-fragments in REGISTERS
// (loop-invariant across the whole integration); only the 16x256 activation
// ping-pong buffers live in LDS. State + RK4 accumulators are fp32 registers.
// Matmuls: v_wmma_f32_16x16x32_f16 (f16 in, fp32 accumulate).

#define PA 264   // f16 pitch for activation rows (528 B: 16B-aligned, bank-spread)

constexpr int   DD     = 256;          // state dim (= hidden dim)
constexpr int   BB     = 4096;         // batch
constexpr int   STEPS  = 100;
constexpr float DT     = 0.01f;
constexpr size_t TRAJ_STRIDE = (size_t)BB * DD;   // floats per trajectory slice

typedef __attribute__((ext_vector_type(16))) _Float16 v16h;
typedef __attribute__((ext_vector_type(8)))  _Float16 v8h;
typedef __attribute__((ext_vector_type(8)))  float    v8f;

union Frag16 { v16h v; v8h h[2]; };

// A-matrix fragment, 16x32 f16 (ISA 7.12.2): lane&15 = row M;
// lanes 0-15: K = kb+{0..7, 16..23}; lanes 16-31: K = kb+{8..15, 24..31}.
__device__ __forceinline__ v16h load_fragA(const _Float16* base, int lane, int kb) {
    const int m   = lane & 15;
    const int off = (lane >> 4) << 3;            // 0 or 8
    const _Float16* p = base + m * PA + kb + off;
    Frag16 f;
    f.h[0] = *(const v8h*)(p);                   // K = kb+off+0..7   (ds_load_b128)
    f.h[1] = *(const v8h*)(p + 16);              // K = kb+off+16..23 (ds_load_b128)
    return f.v;
}

// B-matrix fragment, 32x16 f16, loaded straight from global W[k][n] (k-major):
// lane&15 = col N; lanes 0-15: K = kb+{0..7, 16..23}; lanes 16-31: K = kb+{8..15, 24..31}.
// Done ONCE at kernel start; fragments stay in VGPRs for all 100 steps.
__device__ __forceinline__ v16h load_fragB_global(const float* __restrict__ W,
                                                  int lane, int ntile, int kb) {
    const int n   = ntile * 16 + (lane & 15);
    const int off = (lane >> 4) << 3;            // 0 or 8
    const float* p = W + (size_t)(kb + off) * DD + n;
    v16h f;
    #pragma unroll
    for (int e = 0; e < 16; ++e)
        f[e] = (_Float16)p[(size_t)(e + ((e >> 3) << 3)) * DD];
    return f;
}

__global__ __launch_bounds__(256)
void node_rk4_kernel(const float* __restrict__ x0,
                     const float* __restrict__ tspan,
                     const float* __restrict__ W1,
                     const float* __restrict__ b1,
                     const float* __restrict__ W2,
                     const float* __restrict__ b2,
                     float* __restrict__ out)
{
    // LDS: just the two activation buffers (~16.5 KB)
    __shared__ alignas(16) _Float16 sA[16 * PA];    // stage input (f16)
    __shared__ alignas(16) _Float16 sH[16 * PA];    // tanh activations (f16)

    const int tid  = threadIdx.x;
    const int lane = tid & 31;
    const int wave = tid >> 5;
    const int row0 = blockIdx.x * 16;              // this block's batch-row base

    // ---- times output (out[0..100]) ----
    const float t0 = tspan[0];
    if (blockIdx.x == 0 && tid <= STEPS) out[tid] = t0 + DT * (float)tid;
    float* traj = out + (STEPS + 1);

    // ---- static lane ownership (matches WMMA C/D layout) ----
    const int nn0   = (wave * 2 + 0) * 16 + (lane & 15);
    const int nn1   = (wave * 2 + 1) * 16 + (lane & 15);
    const int mbase = (lane >> 4) ? 8 : 0;

    // ---- biases for owned columns -> registers ----
    const float b1n0 = b1[nn0], b1n1 = b1[nn1];
    const float b2n0 = b2[nn0], b2n1 = b2[nn1];

    // ---- weight fragments -> registers (invariant for whole kernel) ----
    v16h wf1[2][8], wf2[2][8];
    #pragma unroll
    for (int t = 0; t < 2; ++t)
        #pragma unroll
        for (int kbi = 0; kbi < 8; ++kbi) {
            wf1[t][kbi] = load_fragB_global(W1, lane, wave * 2 + t, kbi * 32);
            wf2[t][kbi] = load_fragB_global(W2, lane, wave * 2 + t, kbi * 32);
        }

    float xr0[8], xr1[8];      // ODE state, fp32, in registers
    float ar0[8], ar1[8];      // RK4 accumulator k1+2k2+2k3+k4

    // ---- init: state regs, stage-input LDS, trajectory[0] ----
    #pragma unroll
    for (int r = 0; r < 8; ++r) {
        const int m = mbase + r;
        const float v0 = x0[(size_t)(row0 + m) * DD + nn0];
        const float v1 = x0[(size_t)(row0 + m) * DD + nn1];
        xr0[r] = v0; xr1[r] = v1;
        sA[m * PA + nn0] = (_Float16)v0;
        sA[m * PA + nn1] = (_Float16)v1;
        __builtin_nontemporal_store(v0, &traj[(size_t)(row0 + m) * DD + nn0]);
        __builtin_nontemporal_store(v1, &traj[(size_t)(row0 + m) * DD + nn1]);
    }

    const float half  = DT * 0.5f;
    const float sixth = DT / 6.0f;

    for (int step = 0; step < STEPS; ++step) {
        #pragma unroll 1
        for (int stage = 0; stage < 4; ++stage) {
            __syncthreads();   // sA ready for all waves

            // ---- GEMM1: Hpre = A @ W1 ; H = tanh(Hpre + b1) -> sH (f16) ----
            v8f acc0 = {}, acc1 = {};
            #pragma unroll
            for (int kbi = 0; kbi < 8; ++kbi) {
                const v16h fa = load_fragA(sA, lane, kbi * 32);
                acc0 = __builtin_amdgcn_wmma_f32_16x16x32_f16(false, fa, false, wf1[0][kbi],
                                                              (short)0, acc0, false, false);
                acc1 = __builtin_amdgcn_wmma_f32_16x16x32_f16(false, fa, false, wf1[1][kbi],
                                                              (short)0, acc1, false, false);
            }
            #pragma unroll
            for (int r = 0; r < 8; ++r) {
                const int m = mbase + r;
                sH[m * PA + nn0] = (_Float16)tanhf(acc0[r] + b1n0);
                sH[m * PA + nn1] = (_Float16)tanhf(acc1[r] + b1n1);
            }
            __syncthreads();   // sH ready

            // ---- GEMM2: K = H @ W2 + b2 ; fused RK4 stage update ----
            acc0 = (v8f){}; acc1 = (v8f){};
            #pragma unroll
            for (int kbi = 0; kbi < 8; ++kbi) {
                const v16h fa = load_fragA(sH, lane, kbi * 32);
                acc0 = __builtin_amdgcn_wmma_f32_16x16x32_f16(false, fa, false, wf2[0][kbi],
                                                              (short)0, acc0, false, false);
                acc1 = __builtin_amdgcn_wmma_f32_16x16x32_f16(false, fa, false, wf2[1][kbi],
                                                              (short)0, acc1, false, false);
            }

            const float cin = (stage == 2) ? DT : half;           // stage input scale
            const float kw  = (stage == 1 || stage == 2) ? 2.0f : 1.0f;

            #pragma unroll
            for (int r = 0; r < 8; ++r) {
                const int m = mbase + r;
                const float k0 = acc0[r] + b2n0;
                const float k1 = acc1[r] + b2n1;
                if (stage == 0) { ar0[r] = k0;        ar1[r] = k1;        }
                else            { ar0[r] += kw * k0;  ar1[r] += kw * k1;  }

                float nx0, nx1;
                if (stage < 3) {
                    nx0 = xr0[r] + cin * k0;
                    nx1 = xr1[r] + cin * k1;
                } else {
                    xr0[r] += ar0[r] * sixth;  nx0 = xr0[r];
                    xr1[r] += ar1[r] * sixth;  nx1 = xr1[r];
                    float* dst = traj + (size_t)(step + 1) * TRAJ_STRIDE
                                      + (size_t)(row0 + m) * DD;
                    __builtin_nontemporal_store(nx0, dst + nn0);
                    __builtin_nontemporal_store(nx1, dst + nn1);
                }
                sA[m * PA + nn0] = (_Float16)nx0;
                sA[m * PA + nn1] = (_Float16)nx1;
            }
        }
    }
}

extern "C" void kernel_launch(void* const* d_in, const int* in_sizes, int n_in,
                              void* d_out, int out_size, void* d_ws, size_t ws_size,
                              hipStream_t stream) {
    (void)in_sizes; (void)n_in; (void)out_size; (void)d_ws; (void)ws_size;
    const float* x0    = (const float*)d_in[0];
    const float* tspan = (const float*)d_in[1];
    const float* W1    = (const float*)d_in[2];
    const float* b1    = (const float*)d_in[3];
    const float* W2    = (const float*)d_in[4];
    const float* b2    = (const float*)d_in[5];
    float* out = (float*)d_out;

    dim3 grid(BB / 16);   // 256 workgroups, one 16-row slice each
    dim3 block(256);      // 8 wave32 waves
    hipLaunchKernelGGL(node_rk4_kernel, grid, block, 0, stream,
                       x0, tspan, W1, b1, W2, b2, out);
}
